// IAARegression_11038065951256
// MI455X (gfx1250) — compile-verified, matrix operand order checked by reading, not run on hardware
//
#include <hip/hip_runtime.h>
#include <hip/hip_bf16.h>
#include <math.h>

// ===========================================================================
// IAA ViT regression for MI455X (gfx1250, wave32, WMMA + async-to-LDS).
// All GEMMs run on v_wmma_f32_16x16x32_bf16 (f32 accumulate). bf16 tile
// staging uses GLOBAL_LOAD_ASYNC_TO_LDS_B128 (ASYNCcnt) on full tiles,
// writing LDS directly without touching VGPRs; edge tiles fall back to
// branchless clamped vector loads + cndmask-select.
// NOTE: all K dims used here (512, 2048, 64, 912) are multiples of 8.
// Workspace requirement ~300 MB (dominated by [B,H,S,S] bf16 scores).
// ===========================================================================

typedef __bf16 bf16_t;
typedef __attribute__((ext_vector_type(16))) __bf16 v16bf;
typedef __attribute__((ext_vector_type(8)))  __bf16 v8bf;
typedef __attribute__((ext_vector_type(8)))  float  v8f;

#define SEQ     912
#define DMODEL  512
#define NHEAD   8
#define DHEAD   64
#define NB      8
#define NLAYER  6
#define DFF     2048
#define CIN     2048
#define GRID_P  10

#define BM 64
#define BN 64
#define BK 32

__device__ __forceinline__ bf16_t f2b(float f) { return (bf16_t)f; }
__device__ __forceinline__ float  b2f(bf16_t h) { return (float)h; }
__device__ __forceinline__ float  gelu_f(float v) {
  return 0.5f * v * (1.0f + erff(v * 0.70710678118654752f));
}

// 32-bit wave-relative LDS byte address (addrspace(3) representation) for the
// async-DMA instructions, which take the LDS destination in a single VGPR.
typedef const __attribute__((address_space(3))) char* lds_cptr_t;
__device__ __forceinline__ unsigned lds_off(const void* p) {
  return (unsigned)(size_t)(lds_cptr_t)p;
}

// Async copy of 16 bytes global -> LDS (tracked by ASYNCcnt, bypasses VGPRs).
__device__ __forceinline__ void async_copy16(const bf16_t* gsrc,
                                             const bf16_t* ldst) {
  asm volatile("global_load_async_to_lds_b128 %0, %1, off"
               :
               : "v"(lds_off(ldst)), "v"(gsrc)
               : "memory");
}
__device__ __forceinline__ void wait_async0() {
  asm volatile("s_wait_asynccnt 0x0" ::: "memory");
}

// Gather one 16x32 bf16 WMMA operand fragment from an LDS tile row.
// Per the CDNA5 ISA 16-bit operand layout: lanes 0-15 hold K=[0..7,16..23],
// lanes 16-31 hold K=[8..15,24..31] -> two aligned 16B contiguous DS loads.
__device__ __forceinline__ v16bf load_frag(const bf16_t* rowp, int klo) {
  v8bf lo = *(const v8bf*)(rowp + klo);
  v8bf hi = *(const v8bf*)(rowp + 16 + klo);
  return __builtin_shufflevector(lo, hi, 0, 1, 2, 3, 4, 5, 6, 7,
                                 8, 9, 10, 11, 12, 13, 14, 15);
}

// ---------------------------------------------------------------------------
// fp32 -> bf16 conversion (grid-stride)
// ---------------------------------------------------------------------------
__global__ void cvt_f32_bf16(const float* __restrict__ src,
                             bf16_t* __restrict__ dst, long n) {
  long i = (long)blockIdx.x * blockDim.x + threadIdx.x;
  long stride = (long)gridDim.x * blockDim.x;
  for (; i < n; i += stride) dst[i] = f2b(src[i]);
}

// ---------------------------------------------------------------------------
// Generic bf16 WMMA GEMM:  C[m,n] = f( scale * sum_k A[m,k]*B[n,k] + bias[n] )
//   A: row-major [M,K], row stride lda
//   B: !TRANSB -> row-major [N,K], row stride ldb (weights W[N,K] = W^T use)
//       TRANSB -> element (n,k) at B + k*ldb + n  (n contiguous; used for P@V)
//   Two-level batch offsets: off(z) = (z/IC)*outer + (z%IC)*inner
//   EPI: 0 = linear, 1 = GELU.  OutT: float or bf16.
// Block: 128 threads = 4 waves in a 2x2 grid, each wave owns a 32x32 tile
// (4 accumulators), K-step 32 -> 4 x v_wmma_f32_16x16x32_bf16 per step.
// Full tiles stage via async-to-LDS DMA; edges use clamped vector loads.
// ---------------------------------------------------------------------------
template <int EPI, bool TRANSB, typename OutT>
__global__ __launch_bounds__(128) void gemm_bf16_wmma(
    const bf16_t* __restrict__ A, int lda, long aOuter, long aInner,
    const bf16_t* __restrict__ Bm, int ldb, long bOuter, long bInner,
    const float* __restrict__ bias,
    OutT* __restrict__ C, int ldc, long cOuter, long cInner,
    int innerCount, int M, int N, int K, float scale) {
  __shared__ bf16_t As[BM][BK + 8];
  __shared__ bf16_t Bs[BN][BK + 8];

  const int z = blockIdx.z;
  const bf16_t* Ab = A + (long)(z / innerCount) * aOuter + (long)(z % innerCount) * aInner;
  const bf16_t* Bb = Bm + (long)(z / innerCount) * bOuter + (long)(z % innerCount) * bInner;
  OutT* Cb = C + (long)(z / innerCount) * cOuter + (long)(z % innerCount) * cInner;

  const int tileM = blockIdx.y * BM;
  const int tileN = blockIdx.x * BN;
  const int tid = threadIdx.x;
  const int lane = tid & 31;
  const int wave = tid >> 5;
  const int wm = (wave >> 1) * 32;
  const int wn = (wave & 1) * 32;
  const int klo = (lane >> 4) * 8;
  const int rrow = lane & 15;

  const bool fullM = (tileM + BM <= M);  // block-uniform
  const bool fullN = (tileN + BN <= N);  // block-uniform

  const v8bf zv = {};
  v8f acc[2][2] = {};

  for (int k0 = 0; k0 < K; k0 += BK) {
    const bool fullK = (k0 + BK <= K);   // block-uniform
    // ---- stage A tile (64 rows x 32 k) ----
    {
      int r = tid >> 1;
      int kc = (tid & 1) * 16;
      int gm = tileM + r;
      if (fullM && fullK) {
        // async DMA: global -> LDS directly, no VGPR round trip (EXEC all 1s)
        const bf16_t* gp = Ab + (long)gm * lda + k0 + kc;
        async_copy16(gp, &As[r][kc]);
        async_copy16(gp + 8, &As[r][kc + 8]);
      } else {
        int gmc = gm < M ? gm : M - 1;                // clamp row
        const bf16_t* ap = Ab + (long)gmc * lda;
#pragma unroll
        for (int u = 0; u < 2; ++u) {
          int kk = k0 + kc + u * 8;
          bool ok = (gm < M) && (kk + 8 <= K);
          int kkc = (kk + 8 <= K) ? kk : (K - 8);     // K % 8 == 0 -> aligned
          v8bf val = *(const v8bf*)(ap + kkc);
          if (!ok) val = zv;
          *(v8bf*)&As[r][kc + u * 8] = val;
        }
      }
    }
    // ---- stage B tile ----
    if (!TRANSB) {
      int r = tid >> 1;
      int kc = (tid & 1) * 16;
      int gn = tileN + r;
      if (fullN && fullK) {
        const bf16_t* gp = Bb + (long)gn * ldb + k0 + kc;
        async_copy16(gp, &Bs[r][kc]);
        async_copy16(gp + 8, &Bs[r][kc + 8]);
      } else {
        int gnc = gn < N ? gn : N - 1;
        const bf16_t* bp = Bb + (long)gnc * ldb;
#pragma unroll
        for (int u = 0; u < 2; ++u) {
          int kk = k0 + kc + u * 8;
          bool ok = (gn < N) && (kk + 8 <= K);
          int kkc = (kk + 8 <= K) ? kk : (K - 8);
          v8bf val = *(const v8bf*)(bp + kkc);
          if (!ok) val = zv;
          *(v8bf*)&Bs[r][kc + u * 8] = val;
        }
      }
    } else {
      // element (n,k) at Bb + k*ldb + n : vectorize along contiguous n
      int kl = tid >> 2;
      int nc = (tid & 3) * 16;
      int gk = k0 + kl;
      int gkc = gk < K ? gk : K - 1;                  // clamp k row
      const bf16_t* bp = Bb + (long)gkc * ldb;
#pragma unroll
      for (int u = 0; u < 2; ++u) {
        int gn = tileN + nc + u * 8;
        bool ok = (gk < K) && (gn + 8 <= N);
        int gnc = (gn + 8 <= N) ? gn : (N - 8);       // N % 8 == 0 -> aligned
        v8bf val = *(const v8bf*)(bp + gnc);
        if (!ok) val = zv;
#pragma unroll
        for (int e = 0; e < 8; ++e) Bs[nc + u * 8 + e][kl] = val[e];
      }
      if (gk + BK < K)
        __builtin_prefetch(bp + (long)BK * ldb + tileN + nc, 0, 1);
    }
    wait_async0();   // ASYNCcnt == 0 -> our DMA writes landed in LDS
    __syncthreads();

    v16bf afr[2], bfr[2];
#pragma unroll
    for (int i = 0; i < 2; ++i) {
      afr[i] = load_frag(&As[wm + i * 16 + rrow][0], klo);
      bfr[i] = load_frag(&Bs[wn + i * 16 + rrow][0], klo);
    }
#pragma unroll
    for (int i = 0; i < 2; ++i)
#pragma unroll
      for (int j = 0; j < 2; ++j)
        acc[i][j] = __builtin_amdgcn_wmma_f32_16x16x32_bf16(
            false, afr[i], false, bfr[j], (short)0, acc[i][j], false, false);
    __syncthreads();
  }

  // ---- epilogue: C layout lanes 0-15 -> M=r, lanes 16-31 -> M=r+8 ----
#pragma unroll
  for (int i = 0; i < 2; ++i) {
#pragma unroll
    for (int j = 0; j < 2; ++j) {
      int nn = tileN + wn + j * 16 + rrow;
      if (nn >= N) continue;
      float bvv = bias ? bias[nn] : 0.0f;
#pragma unroll
      for (int r = 0; r < 8; ++r) {
        int mm = tileM + wm + i * 16 + r + ((lane >> 4) << 3);
        if (mm >= M) continue;
        float v = acc[i][j][r] * scale + bvv;
        if (EPI == 1) v = gelu_f(v);
        Cb[(long)mm * ldc + nn] = (OutT)v;
      }
    }
  }
}

// ---------------------------------------------------------------------------
// Embed GEMM: x[b, seq_off+p, d] = sum_c feat[b,c,p]*W[d,c] + conv_b[d]
//                                   + chan_emb[d] + pos_emb[ti(p),tj(p),d]
// feat is channel-major -> transpose-stage into LDS with f32->bf16 convert.
// Weight tile (already bf16) staged by async DMA.
// ---------------------------------------------------------------------------
__global__ __launch_bounds__(128) void embed_gemm(
    const float* __restrict__ feat, const bf16_t* __restrict__ Wb,
    const float* __restrict__ conv_b, const float* __restrict__ chan_emb,
    const float* __restrict__ pos_emb, float* __restrict__ x,
    bf16_t* __restrict__ xb, int Hh, int Ww, int seq_off) {
  __shared__ bf16_t As[BM][BK + 8];
  __shared__ bf16_t Bs[BN][BK + 8];
  const int HW = Hh * Ww;
  const int M = HW, N = DMODEL, K = CIN;
  const int b = blockIdx.z;
  const float* Af = feat + (long)b * K * HW;
  const int tileM = blockIdx.y * BM;
  const int tileN = blockIdx.x * BN;
  const int tid = threadIdx.x;
  const int lane = tid & 31;
  const int wave = tid >> 5;
  const int wm = (wave >> 1) * 32;
  const int wn = (wave & 1) * 32;
  const int klo = (lane >> 4) * 8;
  const int rrow = lane & 15;
  const bool fullTile = (tileM + BM <= M) && ((HW & 3) == 0);
  v8f acc[2][2] = {};

  for (int k0 = 0; k0 < K; k0 += BK) {
    {  // A: feat[k][m] (m contiguous), convert to bf16, store transposed
      int kl = tid >> 2;
      int mc = (tid & 3) * 16;
      int gk = k0 + kl;  // K == CIN is a multiple of BK -> always in bounds
      const float* src = Af + (long)gk * HW + tileM + mc;
      if (fullTile) {
#pragma unroll
        for (int e4 = 0; e4 < 4; ++e4) {
          float4 fv = *(const float4*)(src + e4 * 4);
          As[mc + e4 * 4 + 0][kl] = f2b(fv.x);
          As[mc + e4 * 4 + 1][kl] = f2b(fv.y);
          As[mc + e4 * 4 + 2][kl] = f2b(fv.z);
          As[mc + e4 * 4 + 3][kl] = f2b(fv.w);
        }
      } else {
        const float* srow = Af + (long)gk * HW;
#pragma unroll
        for (int e = 0; e < 16; ++e) {
          int gm = tileM + mc + e;
          int gmc = gm < M ? gm : M - 1;              // clamp address
          float fv = srow[gmc];
          As[mc + e][kl] = f2b(gm < M ? fv : 0.0f);   // select, no branch
        }
      }
    }
    {  // B: conv weights bf16 [N=512][K=2048] -> async DMA, always full
      int r = tid >> 1;
      int kc = (tid & 1) * 16;
      const bf16_t* bp = Wb + (long)(tileN + r) * K + k0 + kc;
      async_copy16(bp, &Bs[r][kc]);
      async_copy16(bp + 8, &Bs[r][kc + 8]);
    }
    wait_async0();
    __syncthreads();
    v16bf afr[2], bfr[2];
#pragma unroll
    for (int i = 0; i < 2; ++i) {
      afr[i] = load_frag(&As[wm + i * 16 + rrow][0], klo);
      bfr[i] = load_frag(&Bs[wn + i * 16 + rrow][0], klo);
    }
#pragma unroll
    for (int i = 0; i < 2; ++i)
#pragma unroll
      for (int j = 0; j < 2; ++j)
        acc[i][j] = __builtin_amdgcn_wmma_f32_16x16x32_bf16(
            false, afr[i], false, bfr[j], (short)0, acc[i][j], false, false);
    __syncthreads();
  }

#pragma unroll
  for (int i = 0; i < 2; ++i) {
#pragma unroll
    for (int j = 0; j < 2; ++j) {
      int nn = tileN + wn + j * 16 + rrow;
      float base = conv_b[nn] + chan_emb[nn];
#pragma unroll
      for (int r = 0; r < 8; ++r) {
        int mm = tileM + wm + i * 16 + r + ((lane >> 4) << 3);
        if (mm >= M) continue;
        int pi = (mm / Ww) * GRID_P / Hh;
        int pj = (mm % Ww) * GRID_P / Ww;
        float v = acc[i][j][r] + base + pos_emb[(pi * GRID_P + pj) * DMODEL + nn];
        long rowx = (long)b * SEQ + seq_off + mm;
        x[rowx * DMODEL + nn] = v;
        xb[rowx * DMODEL + nn] = f2b(v);
      }
    }
  }
}

// ---------------------------------------------------------------------------
__global__ void write_cls(const float* __restrict__ cls, float* __restrict__ x,
                          bf16_t* __restrict__ xb) {
  int b = blockIdx.x;
  int d = threadIdx.x;  // 512 threads
  float v = cls[d];
  long idx = (long)b * SEQ * DMODEL + d;
  x[idx] = v;
  xb[idx] = f2b(v);
}

// ---------------------------------------------------------------------------
// Masked softmax, in place on bf16 scores. One wave per row of length SEQ.
// ---------------------------------------------------------------------------
__global__ __launch_bounds__(256) void softmax_mask(bf16_t* __restrict__ sc,
                                                    const int* __restrict__ mask) {
  const int S = SEQ;
  const int wave = threadIdx.x >> 5;
  const int lane = threadIdx.x & 31;
  const long rowi = (long)blockIdx.x * 8 + wave;  // < B*H*S
  const int b = (int)(rowi / ((long)NHEAD * S));
  bf16_t* row = sc + rowi * S;
  const int* mrow = mask + (long)b * S;

  float mx = -3.0e38f;
  for (int j = lane; j < S; j += 32) {
    float v = b2f(row[j]);
    if (mrow[j] == 0) v = -1.0e9f;
    mx = fmaxf(mx, v);
  }
#pragma unroll
  for (int m = 16; m > 0; m >>= 1) mx = fmaxf(mx, __shfl_xor(mx, m, 32));
  float sum = 0.0f;
  for (int j = lane; j < S; j += 32) {
    float v = b2f(row[j]);
    if (mrow[j] == 0) v = -1.0e9f;
    sum += expf(v - mx);
  }
#pragma unroll
  for (int m = 16; m > 0; m >>= 1) sum += __shfl_xor(sum, m, 32);
  float inv = 1.0f / sum;
  for (int j = lane; j < S; j += 32) {
    float v = b2f(row[j]);
    if (mrow[j] == 0) v = -1.0e9f;
    row[j] = f2b(expf(v - mx) * inv);
  }
}

// ---------------------------------------------------------------------------
// Fused residual-add + LayerNorm: x = LN(x + res); also refresh bf16 shadow.
// One 256-thread block (8 waves) per 512-element row.
// ---------------------------------------------------------------------------
__global__ __launch_bounds__(256) void residual_ln(
    float* __restrict__ x, const float* __restrict__ res,
    const float* __restrict__ g, const float* __restrict__ beta,
    bf16_t* __restrict__ xb) {
  __shared__ float red[8];
  const long row = blockIdx.x;
  float* xr = x + row * DMODEL;
  const float* rr = res + row * DMODEL;
  const int tid = threadIdx.x;

  float v0 = xr[tid] + rr[tid];
  float v1 = xr[tid + 256] + rr[tid + 256];

  float s = v0 + v1;
#pragma unroll
  for (int m = 16; m > 0; m >>= 1) s += __shfl_xor(s, m, 32);
  if ((tid & 31) == 0) red[tid >> 5] = s;
  __syncthreads();
  if (tid == 0) {
    float t = 0.0f;
    for (int i = 0; i < 8; ++i) t += red[i];
    red[0] = t * (1.0f / DMODEL);
  }
  __syncthreads();
  const float mean = red[0];
  __syncthreads();

  float d0 = v0 - mean, d1 = v1 - mean;
  s = d0 * d0 + d1 * d1;
#pragma unroll
  for (int m = 16; m > 0; m >>= 1) s += __shfl_xor(s, m, 32);
  if ((tid & 31) == 0) red[tid >> 5] = s;
  __syncthreads();
  if (tid == 0) {
    float t = 0.0f;
    for (int i = 0; i < 8; ++i) t += red[i];
    red[0] = t * (1.0f / DMODEL);
  }
  __syncthreads();
  const float inv = rsqrtf(red[0] + 1e-6f);

  float o0 = d0 * inv * g[tid] + beta[tid];
  float o1 = d1 * inv * g[tid + 256] + beta[tid + 256];
  xr[tid] = o0;
  xr[tid + 256] = o1;
  xb[row * DMODEL + tid] = f2b(o0);
  xb[row * DMODEL + tid + 256] = f2b(o1);
}

// ---------------------------------------------------------------------------
// Head (tiny; fp32 VALU): hm = gelu(cls @ W1^T);  out[b] = hm @ w2
// ---------------------------------------------------------------------------
__global__ void head1(const float* __restrict__ x, const float* __restrict__ w1,
                      float* __restrict__ hm) {
  int n = blockIdx.x * blockDim.x + threadIdx.x;  // 0..1023
  int b = blockIdx.y;
  const float* xr = x + (long)b * SEQ * DMODEL;   // cls row
  const float* wr = w1 + (long)n * DMODEL;
  float s = 0.0f;
  for (int d = 0; d < DMODEL; ++d) s += xr[d] * wr[d];
  hm[b * 1024 + n] = gelu_f(s);
}

__global__ void head2(const float* __restrict__ hm, const float* __restrict__ w2,
                      float* __restrict__ out) {
  __shared__ float red[8];
  int b = blockIdx.x;
  int tid = threadIdx.x;  // 256
  float s = 0.0f;
  for (int n = tid; n < 1024; n += 256) s += hm[b * 1024 + n] * w2[n];
#pragma unroll
  for (int m = 16; m > 0; m >>= 1) s += __shfl_xor(s, m, 32);
  if ((tid & 31) == 0) red[tid >> 5] = s;
  __syncthreads();
  if (tid == 0) {
    float t = 0.0f;
    for (int i = 0; i < 8; ++i) t += red[i];
    out[b] = t;
  }
}

// ===========================================================================
extern "C" void kernel_launch(void* const* d_in, const int* in_sizes, int n_in,
                              void* d_out, int out_size, void* d_ws,
                              size_t ws_size, hipStream_t stream) {
  (void)in_sizes; (void)n_in; (void)out_size; (void)ws_size;
  const float* feat_org = (const float*)d_in[0];
  const float* feat_r1  = (const float*)d_in[1];
  const float* feat_r2  = (const float*)d_in[2];
  const float* conv_w   = (const float*)d_in[3];
  const float* conv_b   = (const float*)d_in[4];
  const float* emb_org  = (const float*)d_in[5];
  const float* emb_r1   = (const float*)d_in[6];
  const float* emb_r2   = (const float*)d_in[7];
  const float* pos_emb  = (const float*)d_in[8];
  const float* cls_tok  = (const float*)d_in[9];
  const float* Wq = (const float*)d_in[10]; const float* bq = (const float*)d_in[11];
  const float* Wk = (const float*)d_in[12]; const float* bk = (const float*)d_in[13];
  const float* Wv = (const float*)d_in[14]; const float* bvp = (const float*)d_in[15];
  const float* Wo = (const float*)d_in[16]; const float* bo = (const float*)d_in[17];
  const float* ln1g = (const float*)d_in[18]; const float* ln1b = (const float*)d_in[19];
  const float* w1 = (const float*)d_in[20]; const float* b1 = (const float*)d_in[21];
  const float* w2 = (const float*)d_in[22]; const float* b2 = (const float*)d_in[23];
  const float* ln2g = (const float*)d_in[24]; const float* ln2b = (const float*)d_in[25];
  const float* pw1 = (const float*)d_in[26]; const float* pw2 = (const float*)d_in[27];
  const int* mask = (const int*)d_in[28];
  float* out = (float*)d_out;

  const int S = SEQ, D = DMODEL;
  const long rows = (long)NB * S;  // 7296 = 114 * 64

  // ---- carve workspace ----
  size_t off = 0;
  auto carve = [&](size_t bytes) -> char* {
    off = (off + 255) & ~(size_t)255;
    char* p = (char*)d_ws + off;
    off += bytes;
    return p;
  };
  bf16_t* convw_b = (bf16_t*)carve((size_t)D * CIN * 2);
  bf16_t* Wq_b = (bf16_t*)carve((size_t)NLAYER * D * D * 2);
  bf16_t* Wk_b = (bf16_t*)carve((size_t)NLAYER * D * D * 2);
  bf16_t* Wv_b = (bf16_t*)carve((size_t)NLAYER * D * D * 2);
  bf16_t* Wo_b = (bf16_t*)carve((size_t)NLAYER * D * D * 2);
  bf16_t* w1_b = (bf16_t*)carve((size_t)NLAYER * DFF * D * 2);
  bf16_t* w2_b = (bf16_t*)carve((size_t)NLAYER * D * DFF * 2);
  float*  x    = (float*)carve((size_t)rows * D * 4);
  bf16_t* xb   = (bf16_t*)carve((size_t)rows * D * 2);
  bf16_t* q    = (bf16_t*)carve((size_t)rows * D * 2);
  bf16_t* kbuf = (bf16_t*)carve((size_t)rows * D * 2);
  bf16_t* vbuf = (bf16_t*)carve((size_t)rows * D * 2);
  bf16_t* sc   = (bf16_t*)carve((size_t)NB * NHEAD * S * S * 2);
  bf16_t* ctx  = (bf16_t*)carve((size_t)rows * D * 2);
  float*  att  = (float*)carve((size_t)rows * D * 4);
  bf16_t* h1   = (bf16_t*)carve((size_t)rows * DFF * 2);
  float*  ff   = (float*)carve((size_t)rows * D * 4);
  float*  hm   = (float*)carve((size_t)NB * 1024 * 4);

  // ---- weights -> bf16 (once per call; deterministic) ----
  auto cvt = [&](const float* s, bf16_t* d, long n) {
    cvt_f32_bf16<<<dim3(1024), dim3(256), 0, stream>>>(s, d, n);
  };
  cvt(conv_w, convw_b, (long)D * CIN);
  cvt(Wq, Wq_b, (long)NLAYER * D * D);
  cvt(Wk, Wk_b, (long)NLAYER * D * D);
  cvt(Wv, Wv_b, (long)NLAYER * D * D);
  cvt(Wo, Wo_b, (long)NLAYER * D * D);
  cvt(w1, w1_b, (long)NLAYER * DFF * D);
  cvt(w2, w2_b, (long)NLAYER * D * DFF);

  // ---- embedding: conv-as-GEMM with fused bias/channel/pos epilogue ----
  embed_gemm<<<dim3(8, 12, NB), dim3(128), 0, stream>>>(
      feat_org, convw_b, conv_b, emb_org, pos_emb, x, xb, 24, 32, 1);
  embed_gemm<<<dim3(8, 2, NB), dim3(128), 0, stream>>>(
      feat_r1, convw_b, conv_b, emb_r1, pos_emb, x, xb, 9, 12, 769);
  embed_gemm<<<dim3(8, 1, NB), dim3(128), 0, stream>>>(
      feat_r2, convw_b, conv_b, emb_r2, pos_emb, x, xb, 5, 7, 877);
  write_cls<<<dim3(NB), dim3(512), 0, stream>>>(cls_tok, x, xb);

  const int mt = (int)(rows / BM);  // 114
  const int st = (S + BM - 1) / BM; // 15
  for (int i = 0; i < NLAYER; ++i) {
    const bf16_t* wq = Wq_b + (size_t)i * D * D;
    const bf16_t* wk = Wk_b + (size_t)i * D * D;
    const bf16_t* wv = Wv_b + (size_t)i * D * D;
    const bf16_t* wo = Wo_b + (size_t)i * D * D;
    const bf16_t* w1i = w1_b + (size_t)i * DFF * D;
    const bf16_t* w2i = w2_b + (size_t)i * D * DFF;

    // Q, K, V projections (bf16 out, [B,S,H*DH])
    gemm_bf16_wmma<0, false, bf16_t><<<dim3(D / BN, mt, 1), dim3(128), 0, stream>>>(
        xb, D, 0, 0, wq, D, 0, 0, bq + i * D, q, D, 0, 0, 1, (int)rows, D, D, 1.0f);
    gemm_bf16_wmma<0, false, bf16_t><<<dim3(D / BN, mt, 1), dim3(128), 0, stream>>>(
        xb, D, 0, 0, wk, D, 0, 0, bk + i * D, kbuf, D, 0, 0, 1, (int)rows, D, D, 1.0f);
    gemm_bf16_wmma<0, false, bf16_t><<<dim3(D / BN, mt, 1), dim3(128), 0, stream>>>(
        xb, D, 0, 0, wv, D, 0, 0, bvp + i * D, vbuf, D, 0, 0, 1, (int)rows, D, D, 1.0f);

    // scores[z=b*8+h] = scale * Q K^T  (batched, K=64)
    gemm_bf16_wmma<0, false, bf16_t><<<dim3(st, st, NB * NHEAD), dim3(128), 0, stream>>>(
        q, D, (long)S * D, DHEAD, kbuf, D, (long)S * D, DHEAD, nullptr,
        sc, S, (long)NHEAD * S * S, (long)S * S, NHEAD, S, S, DHEAD, 0.125f);

    softmax_mask<<<dim3((NB * NHEAD * S) / 8), dim3(256), 0, stream>>>(sc, mask);

    // ctx[z] = P @ V  (V read K-major via TRANSB path; K=912 padded in LDS)
    gemm_bf16_wmma<0, true, bf16_t><<<dim3(1, st, NB * NHEAD), dim3(128), 0, stream>>>(
        sc, S, (long)NHEAD * S * S, (long)S * S, vbuf, D, (long)S * D, DHEAD, nullptr,
        ctx, D, (long)S * D, DHEAD, NHEAD, S, DHEAD, S, 1.0f);

    // output projection (fp32 out for residual)
    gemm_bf16_wmma<0, false, float><<<dim3(D / BN, mt, 1), dim3(128), 0, stream>>>(
        ctx, D, 0, 0, wo, D, 0, 0, bo + i * D, att, D, 0, 0, 1, (int)rows, D, D, 1.0f);

    residual_ln<<<dim3((unsigned)rows), dim3(256), 0, stream>>>(
        x, att, ln1g + i * D, ln1b + i * D, xb);

    // FFN: gelu(x W1^T + b1) W2^T + b2
    gemm_bf16_wmma<1, false, bf16_t><<<dim3(DFF / BN, mt, 1), dim3(128), 0, stream>>>(
        xb, D, 0, 0, w1i, D, 0, 0, b1 + i * DFF, h1, DFF, 0, 0, 1, (int)rows, DFF, D, 1.0f);
    gemm_bf16_wmma<0, false, float><<<dim3(D / BN, mt, 1), dim3(128), 0, stream>>>(
        h1, DFF, 0, 0, w2i, DFF, 0, 0, b2 + i * D, ff, D, 0, 0, 1, (int)rows, D, DFF, 1.0f);

    residual_ln<<<dim3((unsigned)rows), dim3(256), 0, stream>>>(
        x, ff, ln2g + i * D, ln2b + i * D, xb);
  }

  // ---- MLP head on cls rows ----
  head1<<<dim3(4, NB), dim3(256), 0, stream>>>(x, pw1, hm);
  head2<<<dim3(NB), dim3(256), 0, stream>>>(hm, pw2, out);
}
